// GroupedLinear_73804718014878
// MI455X (gfx1250) — matrix operand order repaired and checked by reading, not compiled, not run on hardware
//
#include <hip/hip_runtime.h>

// Grouped GEMM: out[rows_e] = x[rows_e] @ W[e], tokens sorted by expert.
// fp32 in/out; math via bf16x3 split-WMMA (hi*hi + hi*lo + lo*hi) with f32 accum.
// Double-buffered LDS pipeline, clamped (branch-free) ragged-row loads.

#define E_      128
#define KDIM    2048
#define NDIM    1408
#define SDIM    16384
#define BM      128
#define BN      128
#define BK      32
#define ASTR    40          // LDS row stride (bf16 elems), 80B: 16B-aligned frag loads, bank-skewed
#define BSTR    40
#define MAX_TILES 256       // sum_e ceil(bs_e/BM) <= S/BM + E = 128 + 128

typedef __attribute__((ext_vector_type(16))) __bf16 v16bf;
typedef __attribute__((ext_vector_type(8)))  float  v8f;

struct Tile { int e; int row; int rows; };

__device__ __forceinline__ unsigned short f32_to_bf16_rne(float f) {
    unsigned int u = __float_as_uint(f);
    unsigned int r = u + 0x7FFFu + ((u >> 16) & 1u);
    return (unsigned short)(r >> 16);
}

__device__ __forceinline__ void split_f32(float f, unsigned short& hi, unsigned short& lo) {
    hi = f32_to_bf16_rne(f);
    float fh = __uint_as_float((unsigned int)hi << 16);
    lo = f32_to_bf16_rne(f - fh);
}

// Build a 16x16x32 bf16 fragment from two 16B LDS reads (ds_load_b128 x2).
__device__ __forceinline__ v16bf frag16(const unsigned short* p0, const unsigned short* p1) {
    union { uint4 q[2]; v16bf v; } u;
    u.q[0] = *(const uint4*)p0;
    u.q[1] = *(const uint4*)p1;
    return u.v;
}

// Phase 1: scan ragged group sizes into a fixed-size tile table (deterministic grid).
__global__ void build_tiles_kernel(const int* __restrict__ bs, Tile* __restrict__ tiles) {
    if (threadIdx.x == 0 && blockIdx.x == 0) {
        int off = 0, cnt = 0;
        for (int e = 0; e < E_; ++e) {
            int b = bs[e];
            for (int r = 0; r < b; r += BM) {
                if (cnt < MAX_TILES) {
                    tiles[cnt].e = e;
                    tiles[cnt].row = off + r;
                    tiles[cnt].rows = (b - r < BM) ? (b - r) : BM;
                }
                ++cnt;
            }
            off += b;
        }
        for (int q = cnt; q < MAX_TILES; ++q) { tiles[q].e = 0; tiles[q].row = 0; tiles[q].rows = 0; }
    }
}

__global__ __launch_bounds__(256) void grouped_gemm_wmma(
    const float* __restrict__ x, const float* __restrict__ w,
    const Tile* __restrict__ tiles, float* __restrict__ out)
{
    // Double-buffered staging: 2 x (A 10KB + B 10KB) x (hi+lo) = 80 KB.
    __shared__ __align__(16) unsigned short AsHi[2][BM * ASTR];
    __shared__ __align__(16) unsigned short AsLo[2][BM * ASTR];
    __shared__ __align__(16) unsigned short BsHi[2][BN * BSTR];
    __shared__ __align__(16) unsigned short BsLo[2][BN * BSTR];

    const Tile t = tiles[blockIdx.y];
    if (t.rows <= 0) return;                   // block-uniform exit (EXEC stays full for WMMA)

    const int  rows     = t.rows;
    const int  rowStart = t.row;
    const int  nBase    = blockIdx.x * BN;
    const float* __restrict__ xA = x + (long)rowStart * KDIM;
    const float* __restrict__ wB = w + (long)t.e * KDIM * NDIM + nBase;

    const int tid   = threadIdx.x;
    const int lane  = tid & 31;
    const int wave  = tid >> 5;
    const int wm    = wave & 3;                // 4 waves along M: 32 rows each
    const int wn    = wave >> 2;               // 2 waves along N: 64 cols each
    const int l16   = lane & 15;
    const int lhalf = lane >> 4;

    // Staging coordinates (per thread, loop-invariant).
    const int ak4 = (tid & 7) * 4;             // A: k offset within slice
    const int amr = tid >> 3;                  // A: row within 32-row pass
    const int bn4 = (tid & 31) * 4;            // B: n offset
    const int bkr = tid >> 5;                  // B: k row within 8-row pass
    // Clamped A rows: OOB rows replicate a valid row; their outputs are never stored.
    int amc[4];
#pragma unroll
    for (int p = 0; p < 4; ++p) {
        int m = p * 32 + amr;
        amc[p] = (m < rows) ? m : (rows - 1);
    }

    v8f acc[2][4] = {};
    float4 ra[4], rb[4];

    // Prologue: load K-slice 0 (8 x global_load_b128, unconditional -> clause-able).
#pragma unroll
    for (int p = 0; p < 4; ++p)
        ra[p] = *(const float4*)(xA + (long)amc[p] * KDIM + ak4);
#pragma unroll
    for (int p = 0; p < 4; ++p)
        rb[p] = *(const float4*)(wB + (long)(p * 8 + bkr) * NDIM + bn4);

    int buf = 0;
    for (int kk = 0; kk < KDIM; kk += BK) {
        // ---- stage held registers -> LDS[buf] (hi/lo split) ----
#pragma unroll
        for (int p = 0; p < 4; ++p) {
            unsigned short h0, h1, h2, h3, l0, l1, l2, l3;
            split_f32(ra[p].x, h0, l0); split_f32(ra[p].y, h1, l1);
            split_f32(ra[p].z, h2, l2); split_f32(ra[p].w, h3, l3);
            const int o = (p * 32 + amr) * ASTR + ak4;       // 8B-aligned -> ds_store_b64
            *(uint2*)&AsHi[buf][o] = make_uint2((unsigned)h0 | ((unsigned)h1 << 16),
                                                (unsigned)h2 | ((unsigned)h3 << 16));
            *(uint2*)&AsLo[buf][o] = make_uint2((unsigned)l0 | ((unsigned)l1 << 16),
                                                (unsigned)l2 | ((unsigned)l3 << 16));
        }
#pragma unroll
        for (int p = 0; p < 4; ++p) {
            const int k = p * 8 + bkr;
            unsigned short h, l;
            split_f32(rb[p].x, h, l); BsHi[buf][(bn4 + 0) * BSTR + k] = h; BsLo[buf][(bn4 + 0) * BSTR + k] = l;
            split_f32(rb[p].y, h, l); BsHi[buf][(bn4 + 1) * BSTR + k] = h; BsLo[buf][(bn4 + 1) * BSTR + k] = l;
            split_f32(rb[p].z, h, l); BsHi[buf][(bn4 + 2) * BSTR + k] = h; BsLo[buf][(bn4 + 2) * BSTR + k] = l;
            split_f32(rb[p].w, h, l); BsHi[buf][(bn4 + 3) * BSTR + k] = h; BsLo[buf][(bn4 + 3) * BSTR + k] = l;
        }
        __syncthreads();

        // ---- issue next K-slice's global loads; consumed at next iteration's store,
        //      so their latency hides behind the 24 WMMAs below ----
        const int kn = kk + BK;
        if (kn < KDIM) {
#pragma unroll
            for (int p = 0; p < 4; ++p)
                ra[p] = *(const float4*)(xA + (long)amc[p] * KDIM + kn + ak4);
#pragma unroll
            for (int p = 0; p < 4; ++p)
                rb[p] = *(const float4*)(wB + (long)(kn + p * 8 + bkr) * NDIM + bn4);
            if (kn + BK < KDIM)                 // pull W one more slice toward L2
                __builtin_prefetch(wB + (long)(kn + BK) * NDIM + bn4, 0, 0);
        }

        // ---- fragments per ISA 16-bit A(16x32)/B(32x16) layouts ----
        // A lane<16: K{0..7,16..23}; lane>=16: K{8..15,24..31}
        // B lane<16: K{0..7,16..23}; lane>=16: K{8..15,24..31}
        const int aK = lhalf ? 8 : 0;
        const int bK = lhalf ? 8 : 0;
        v16bf aH[2], aL[2], bH[4], bL[4];
#pragma unroll
        for (int mf = 0; mf < 2; ++mf) {
            const int o = (wm * 32 + mf * 16 + l16) * ASTR + aK;
            aH[mf] = frag16(&AsHi[buf][o], &AsHi[buf][o + 16]);
            aL[mf] = frag16(&AsLo[buf][o], &AsLo[buf][o + 16]);
        }
#pragma unroll
        for (int nf = 0; nf < 4; ++nf) {
            const int o = (wn * 64 + nf * 16 + l16) * BSTR + bK;
            bH[nf] = frag16(&BsHi[buf][o], &BsHi[buf][o + 16]);
            bL[nf] = frag16(&BsLo[buf][o], &BsLo[buf][o + 16]);
        }

        // ---- 24 x v_wmma_f32_16x16x32_bf16 per wave per K-step ----
#pragma unroll
        for (int mf = 0; mf < 2; ++mf)
#pragma unroll
            for (int nf = 0; nf < 4; ++nf) {
                acc[mf][nf] = __builtin_amdgcn_wmma_f32_16x16x32_bf16(
                    false, aH[mf], false, bH[nf], (short)0, acc[mf][nf], false, false);
                acc[mf][nf] = __builtin_amdgcn_wmma_f32_16x16x32_bf16(
                    false, aH[mf], false, bL[nf], (short)0, acc[mf][nf], false, false);
                acc[mf][nf] = __builtin_amdgcn_wmma_f32_16x16x32_bf16(
                    false, aL[mf], false, bH[nf], (short)0, acc[mf][nf], false, false);
            }
        buf ^= 1;                              // 2 buffers + 1 barrier/iter: max skew 1 iter, safe
    }

    // ---- epilogue: C/D layout -> global fp32 (VGPR r: lanes0-15 M=r, lanes16-31 M=r+8) ----
#pragma unroll
    for (int mf = 0; mf < 2; ++mf)
#pragma unroll
        for (int nf = 0; nf < 4; ++nf) {
            union { v8f v; float f[8]; } r;
            r.v = acc[mf][nf];
            const int ncol = nBase + wn * 64 + nf * 16 + l16;
#pragma unroll
            for (int e = 0; e < 8; ++e) {
                const int m = wm * 32 + mf * 16 + (lhalf ? e + 8 : e);
                if (m < rows)
                    out[(long)(rowStart + m) * NDIM + ncol] = r.f[e];
            }
        }
}

extern "C" void kernel_launch(void* const* d_in, const int* in_sizes, int n_in,
                              void* d_out, int out_size, void* d_ws, size_t ws_size,
                              hipStream_t stream) {
    const float* x  = (const float*)d_in[0];
    const float* w  = (const float*)d_in[1];
    const int*   bs = (const int*)d_in[2];
    float* out = (float*)d_out;
    Tile* tiles = (Tile*)d_ws;                 // 256 * 12 B = 3 KB scratch

    build_tiles_kernel<<<1, 32, 0, stream>>>(bs, tiles);
    dim3 grid(NDIM / BN, MAX_TILES);           // fixed grid: capture-safe; empty tiles exit
    grouped_gemm_wmma<<<grid, 256, 0, stream>>>(x, w, tiles, out);
}
